// LocalNodeAttentionHeadSumV1_32959579030114
// MI455X (gfx1250) — compile-verified, matrix operand order checked by reading, not run on hardware
//
#include <hip/hip_runtime.h>

// ---------------------------------------------------------------------------
// LocalNodeAttentionHeadSum for MI455X (gfx1250, wave32, WMMA + TDM)
//
// Shapes: B=32, C_IN=1024, T=7, H=W=14 (HW=196), C_INTER=512, t_mid=3
// Algebraic restructuring (exact):
//   y    = xbar @ Wv + bv,   xbar[m,c] = sum_t alpha[m,t] * x[b,c,t,hw]
//   alpha= softmax_t( x_item @ (Wq @ key^T) + bq.key^T )
// Heavy work = two bf16 WMMA GEMMs: [6272,1024]x[1024,512], [6272,512]x[512,1024]
// GEMMs stage the shared B panel into LDS via the Tensor Data Mover
// (tensor_load_to_lds, TENSORcnt), double-buffered against the WMMA pipeline.
// ---------------------------------------------------------------------------

typedef __bf16 bf16;
typedef __bf16 v16bf __attribute__((ext_vector_type(16)));
typedef __bf16 v8bf  __attribute__((ext_vector_type(8)));
typedef float  v8f   __attribute__((ext_vector_type(8)));
typedef unsigned int v4u __attribute__((ext_vector_type(4)));
typedef int v8i __attribute__((ext_vector_type(8)));
typedef int v4i __attribute__((ext_vector_type(4)));

#define DEVINL static __device__ __forceinline__

constexpr int B_   = 32;
constexpr int C_IN = 1024;
constexpr int T_   = 7;
constexpr int HW   = 196;      // 14*14
constexpr int C_I  = 512;      // C_INTER
constexpr int MPIX = B_ * HW;  // 6272
constexpr int TMID = 3;

constexpr int KCH  = 128;      // K elements staged per TDM transfer
constexpr int KROW = KCH + 8;  // LDS row stride (elements): 272B -> bank-conflict-free

// --- workspace layout (bytes, 256-aligned) ---
constexpr size_t alignup(size_t x) { return (x + 255) & ~size_t(255); }
constexpr size_t OFF_KEY  = 0;                                        // 7*512 f32
constexpr size_t OFF_Q0   = alignup(OFF_KEY  + size_t(T_)*C_I*4);     // 7 f32
constexpr size_t OFF_P    = alignup(OFF_Q0   + size_t(T_)*4);         // 1024*7 f32
constexpr size_t OFF_ALPH = alignup(OFF_P    + size_t(C_IN)*T_*4);    // 6272*7 f32
constexpr size_t OFF_WVT  = alignup(OFF_ALPH + size_t(MPIX)*T_*4);    // [512][1024] bf16
constexpr size_t OFF_WOT  = alignup(OFF_WVT  + size_t(C_I)*C_IN*2);   // [1024][512] bf16
constexpr size_t OFF_XBAR = alignup(OFF_WOT  + size_t(C_IN)*C_I*2);   // [6272][1024] bf16
constexpr size_t OFF_Y    = alignup(OFF_XBAR + size_t(MPIX)*C_IN*2);  // [6272][512] bf16

// ---------------------------------------------------------------------------
// key[t,d] = nodes[t,:] . Wk[:,d] + bk[d];  q0[t] = bq . key[t,:]
// ---------------------------------------------------------------------------
__global__ void k_key(const float* __restrict__ nodes, const float* __restrict__ Wk,
                      const float* __restrict__ bk, const float* __restrict__ bq,
                      float* __restrict__ key, float* __restrict__ q0) {
  __shared__ float kl[T_ * C_I];
  const int d = threadIdx.x;  // 0..511
  float acc[T_];
#pragma unroll
  for (int t = 0; t < T_; ++t) acc[t] = bk[d];
  for (int c = 0; c < C_I; ++c) {
    float wk = Wk[c * C_I + d];
#pragma unroll
    for (int t = 0; t < T_; ++t) acc[t] += nodes[t * C_I + c] * wk;
  }
#pragma unroll
  for (int t = 0; t < T_; ++t) {
    key[t * C_I + d] = acc[t];
    kl[t * C_I + d]  = acc[t];
  }
  __syncthreads();
  if (d < T_) {
    float s = 0.f;
    for (int j = 0; j < C_I; ++j) s += bq[j] * kl[d * C_I + j];
    q0[d] = s;
  }
}

// ---------------------------------------------------------------------------
// P[c,t] = Wq[c,:] . key[t,:]   (1024*7 threads)
// ---------------------------------------------------------------------------
__global__ void k_P(const float* __restrict__ Wq, const float* __restrict__ key,
                    float* __restrict__ P) {
  const int i = blockIdx.x * blockDim.x + threadIdx.x;  // < 7168
  const int c = i / T_, t = i - c * T_;
  const float* wr = Wq + (size_t)c * C_I;
  const float* kr = key + (size_t)t * C_I;
  float s = 0.f;
  for (int d = 0; d < C_I; ++d) s += wr[d] * kr[d];
  P[c * T_ + t] = s;
}

// ---------------------------------------------------------------------------
// bf16 transposed weight copies: WvT[n][k] = Wv[k][n];  WoT[n][k] = Wo[k][n]
// ---------------------------------------------------------------------------
__global__ void k_convert(const float* __restrict__ Wv, const float* __restrict__ Wo,
                          bf16* __restrict__ WvT, bf16* __restrict__ WoT) {
  const int i = blockIdx.x * blockDim.x + threadIdx.x;  // < 2*524288
  if (i < C_I * C_IN) {                 // WvT: n<512, k<1024
    int n = i >> 10, k = i & 1023;
    WvT[i] = (bf16)Wv[(size_t)k * C_I + n];
  } else {
    int j = i - C_I * C_IN;             // WoT: n<1024, k<512
    int n = j >> 9, k = j & 511;
    WoT[j] = (bf16)Wo[(size_t)k * C_IN + n];
  }
}

// ---------------------------------------------------------------------------
// alpha[m,t] = softmax_t( sum_c x_item[m,c]*P[c,t] + q0[t] ), one pixel/thread
// ---------------------------------------------------------------------------
__global__ void k_alpha(const float* __restrict__ x, const float* __restrict__ P,
                        const float* __restrict__ q0, float* __restrict__ alpha) {
  __shared__ float Pl[C_IN * T_];  // 28 KB
  for (int i = threadIdx.x; i < C_IN * T_; i += blockDim.x) Pl[i] = P[i];
  __syncthreads();
  const int m = blockIdx.x * blockDim.x + threadIdx.x;
  if (m >= MPIX) return;
  const int b = m / HW, hw = m - b * HW;
  const float* xp = x + (size_t)b * C_IN * T_ * HW + (size_t)TMID * HW + hw;
  float a[T_];
#pragma unroll
  for (int t = 0; t < T_; ++t) a[t] = q0[t];
  for (int c = 0; c < C_IN; ++c) {
    float xv = xp[(size_t)c * T_ * HW];
#pragma unroll
    for (int t = 0; t < T_; ++t) a[t] += xv * Pl[c * T_ + t];
  }
  float mx = a[0];
#pragma unroll
  for (int t = 1; t < T_; ++t) mx = fmaxf(mx, a[t]);
  float s = 0.f;
#pragma unroll
  for (int t = 0; t < T_; ++t) { a[t] = __expf(a[t] - mx); s += a[t]; }
  const float inv = 1.f / s;
  float* ap = alpha + (size_t)m * T_;
#pragma unroll
  for (int t = 0; t < T_; ++t) ap[t] = a[t] * inv;
}

// ---------------------------------------------------------------------------
// xbar[m,c] = sum_t alpha[m,t] * x[b,c,t,hw]  (bf16 out, coalesced over hw)
// ---------------------------------------------------------------------------
__global__ void k_xbar(const float* __restrict__ x, const float* __restrict__ alpha,
                       bf16* __restrict__ xbar) {
  const unsigned i = blockIdx.x * blockDim.x + threadIdx.x;  // < 6272*1024
  const int hw = i % HW;
  const unsigned r = i / HW;
  const int c = r % C_IN;
  const int b = r / C_IN;
  const int m = b * HW + hw;
  const float* ap = alpha + (size_t)m * T_;
  const float* xp = x + ((size_t)b * C_IN + c) * T_ * HW + hw;
  float s = 0.f;
#pragma unroll
  for (int t = 0; t < T_; ++t) s += ap[t] * xp[(size_t)t * HW];
  xbar[(size_t)m * C_IN + c] = (bf16)s;
}

// ---------------------------------------------------------------------------
// Load one 16x32 bf16 operand tile (per-lane: row = lane%16, two 16B K-chunks
// at kh*8 and 16+kh*8, kh = lane/16) -- matches ISA §7.12.2 layout.
// ---------------------------------------------------------------------------
DEVINL v16bf ld_tile(const bf16* p) {
  v8bf lo = *(const v8bf*)p;
  v8bf hi = *(const v8bf*)(p + 16);
  v16bf r;
#pragma unroll
  for (int i = 0; i < 8; ++i) { r[i] = lo[i]; r[i + 8] = hi[i]; }
  return r;
}

// ---------------------------------------------------------------------------
// Issue a TDM descriptor: load a 64-row x KCH-col bf16 tile of Bt[NTOT][K]
// (rows starting at n0, cols at kstage*KCH) into LDS at lds_addr, padding
// 4 DWORDs after every 64 DWORDs (one row) -> LDS row stride = KROW elements.
// D# layout per ISA §8.3/§8.4 (group0: count/lds/global/type; group1:
// data_size, pad, tensor dims, tile dims, dim0 stride).
// This toolchain exposes the 6-arg builtin:
//   (uint32x4 g0, int32x8 g1, int32x4 g2, int32x4 g3, int32x8 gx, i32 cpol)
// ---------------------------------------------------------------------------
template <int K, int NTOT>
DEVINL void tdm_load_b_panel(const bf16* gaddr, unsigned lds_addr) {
  unsigned long long ga = (unsigned long long)(uintptr_t)gaddr;
  v4u g0;
  g0.x = 1u;                                      // count=1, user descriptor
  g0.y = lds_addr;                                // LDS byte address
  g0.z = (unsigned)ga;                            // global_addr[31:0]
  g0.w = (unsigned)(ga >> 32) | (2u << 30);       // global_addr[56:32] | type=2
  v8i g1;
  g1[0] = (int)((1u << 16) |                      // data_size = 2 bytes
                (1u << 20) |                      // pad_enable
                (5u << 22) |                      // pad_interval: 64 DWORDs (one row)
                (3u << 25));                      // pad_amount: 4 DWORDs (8 elems)
  g1[1] = (int)(((unsigned)K & 0xffffu) << 16);   // tensor_dim0[15:0]
  g1[2] = (int)(((unsigned)K >> 16) | (((unsigned)NTOT & 0xffffu) << 16)); // dim0 hi | dim1 lo
  g1[3] = (int)(((unsigned)NTOT >> 16) | ((unsigned)KCH << 16));           // dim1 hi | tile_dim0
  g1[4] = 64;                                     // tile_dim1 = 64 rows; tile_dim2 = 0
  g1[5] = K;                                      // tensor_dim0_stride[31:0]
  g1[6] = 0;                                      // stride hi | dim1_stride lo
  g1[7] = 0;
  v4i z4 = {0, 0, 0, 0};                          // 2D tensor: groups 2/3 unused
  v8i z8 = {0, 0, 0, 0, 0, 0, 0, 0};
  __builtin_amdgcn_tensor_load_to_lds(g0, g1, z4, z4, z8, 0);
}

// ---------------------------------------------------------------------------
// WMMA GEMM: C[M,NTOT] = A[M,K](bf16,row-major) * Bt[NTOT,K]^T (bf16) + bias
// Block = 8 waves = 128M x 64N tile. B panel (64 x KCH) staged in LDS by the
// TDM, double-buffered; TENSORcnt in-order completion => wait<=1 releases the
// previous stage. A rows stream from global (b128), overlapping the TDM.
// OUTF=0: bf16 row-major [M,NTOT]. OUTF=1: f32 scatter to (B,C_IN,1,H,W).
// ---------------------------------------------------------------------------
template <int K, int NTOT, int OUTF>
__global__ void __launch_bounds__(256) k_gemm(const bf16* __restrict__ A,
                                              const bf16* __restrict__ Bt,
                                              const float* __restrict__ bias,
                                              bf16* __restrict__ outB,
                                              float* __restrict__ outF) {
  __shared__ bf16 Bs[2][64 * KROW];  // 2 x 17 KB, padded rows

  const int lane = threadIdx.x & 31;
  const int wave = threadIdx.x >> 5;            // 8 waves
  const int m0 = blockIdx.x * 128 + wave * 16;
  const int n0 = blockIdx.y * 64;
  const int lr = lane & 15;
  const int kh = (lane >> 4) * 8;

  const bf16* arow  = A + (size_t)(m0 + lr) * K + kh;
  const bf16* bpanel = Bt + (size_t)n0 * K;     // 64 rows starting at n0
  const unsigned lds0 = (unsigned)(uintptr_t)&Bs[0][0];
  const unsigned lds1 = (unsigned)(uintptr_t)&Bs[1][0];

  v8f acc[4];
#pragma unroll
  for (int j = 0; j < 4; ++j)
#pragma unroll
    for (int r = 0; r < 8; ++r) acc[j][r] = 0.f;

  constexpr int S = K / KCH;                    // TDM stages

  if (wave == 0) tdm_load_b_panel<K, NTOT>(bpanel, lds0);

  for (int s = 0; s < S; ++s) {
    if (wave == 0) {
      if (s + 1 < S) {
        tdm_load_b_panel<K, NTOT>(bpanel + (size_t)(s + 1) * KCH,
                                  ((s + 1) & 1) ? lds1 : lds0);
        __builtin_amdgcn_s_wait_tensorcnt((short)1);  // stage s landed
      } else {
        __builtin_amdgcn_s_wait_tensorcnt((short)0);
      }
    }
    __syncthreads();

    const bf16* bp = &Bs[s & 1][0];
    // preload all A chunks for this stage (overlaps with LDS reads below)
    v16bf a[KCH / 32];
#pragma unroll
    for (int ks = 0; ks < KCH / 32; ++ks) a[ks] = ld_tile(arow + s * KCH + ks * 32);

#pragma unroll
    for (int ks = 0; ks < KCH / 32; ++ks) {
#pragma unroll
      for (int j = 0; j < 4; ++j) {
        v16bf bm = ld_tile(bp + (j * 16 + lr) * KROW + ks * 32 + kh);
        acc[j] = __builtin_amdgcn_wmma_f32_16x16x32_bf16(
            /*neg_a=*/false, a[ks], /*neg_b=*/false, bm,
            /*c_mod=*/(short)0, acc[j], /*reuse_a=*/false, /*reuse_b=*/false);
      }
    }
    __syncthreads();  // before buffer (s&1) is overwritten by stage s+2
  }

#pragma unroll
  for (int j = 0; j < 4; ++j) {
    const int n = n0 + j * 16 + lr;
    const float bvv = bias[n];
#pragma unroll
    for (int r = 0; r < 8; ++r) {
      const int m = m0 + r + ((lane >> 4) << 3);  // C/D layout: M = r + 8*(lane/16)
      const float v = acc[j][r] + bvv;
      if (OUTF) {
        const int b = m / HW, hw = m - b * HW;
        outF[(size_t)b * (C_IN * HW) + (size_t)n * HW + hw] = v;
      } else {
        outB[(size_t)m * NTOT + n] = (bf16)v;
      }
    }
  }
}

// ---------------------------------------------------------------------------
extern "C" void kernel_launch(void* const* d_in, const int* in_sizes, int n_in,
                              void* d_out, int out_size, void* d_ws, size_t ws_size,
                              hipStream_t stream) {
  const float* x     = (const float*)d_in[0];
  const float* nodes = (const float*)d_in[1];
  const float* Wq    = (const float*)d_in[2];
  const float* bq    = (const float*)d_in[3];
  const float* Wk    = (const float*)d_in[4];
  const float* bk    = (const float*)d_in[5];
  const float* Wv    = (const float*)d_in[6];
  const float* bv    = (const float*)d_in[7];
  const float* Wo    = (const float*)d_in[8];
  const float* bo    = (const float*)d_in[9];

  char* ws = (char*)d_ws;
  float* key   = (float*)(ws + OFF_KEY);
  float* q0    = (float*)(ws + OFF_Q0);
  float* P     = (float*)(ws + OFF_P);
  float* alpha = (float*)(ws + OFF_ALPH);
  bf16*  WvT   = (bf16*)(ws + OFF_WVT);
  bf16*  WoT   = (bf16*)(ws + OFF_WOT);
  bf16*  xbar  = (bf16*)(ws + OFF_XBAR);
  bf16*  y     = (bf16*)(ws + OFF_Y);
  float* out   = (float*)d_out;

  // 1) key (7x512) + q0
  k_key<<<1, 512, 0, stream>>>(nodes, Wk, bk, bq, key, q0);
  // 2) P = Wq @ key^T  (1024x7)
  k_P<<<(C_IN * T_) / 256, 256, 0, stream>>>(Wq, key, P);
  // 3) bf16 transposed weights
  k_convert<<<(2 * C_I * C_IN) / 256, 256, 0, stream>>>(Wv, Wo, WvT, WoT);
  // 4) alpha = softmax_t(x_item @ P + q0)
  k_alpha<<<(MPIX + 255) / 256, 256, 0, stream>>>(x, P, q0, alpha);
  // 5) xbar = sum_t alpha_t * x_t   (reads the full 180 MB once, coalesced)
  k_xbar<<<(MPIX * C_IN) / 256, 256, 0, stream>>>(x, alpha, xbar);
  // 6) y = xbar @ Wv + bv   -> bf16 [6272,512]
  k_gemm<C_IN, C_I, 0><<<dim3(MPIX / 128, C_I / 64), 256, 0, stream>>>(
      xbar, WvT, bv, y, nullptr);
  // 7) out = y @ Wo + bo    -> f32 scattered to (B, C_IN, 1, H, W)
  k_gemm<C_I, C_IN, 1><<<dim3(MPIX / 128, C_IN / 64), 256, 0, stream>>>(
      y, WoT, bo, nullptr, out);
  (void)in_sizes; (void)n_in; (void)out_size; (void)ws_size;
}
